// InnerRNN_63720134804209
// MI455X (gfx1250) — matrix operand order repaired and checked by reading, not compiled, not run on hardware
//
#include <hip/hip_runtime.h>
#include <hip/hip_bf16.h>
#include <math.h>

#define KGRP 8
#define HDIM 250
#define NBATCH 48
#define BKROWS (NBATCH * KGRP)   // 384
#define PROWS  (BKROWS * 7)      // 2688 pair rows
#define EPSLN 1e-5f

#define HP   256                 // HDIM padded
#define A1P  128                 // att1 out padded (100)
#define PKP  512                 // pair K padded (500)
#define TKP  1024                // total K padded (1012)

typedef __attribute__((ext_vector_type(16))) _Float16 v16h;
typedef __attribute__((ext_vector_type(8)))  _Float16 v8h;
typedef __attribute__((ext_vector_type(8)))  float    v8f;

// ---------------------------------------------------------------------------
// WMMA GEMM: C[M,Npad](f32) = A[M,Kpad](f16) @ Bt[Npad,Kpad](f16)^T + bias
// Preconditions: M%64==0, Npad%64==0, Kpad%32==0, operands zero-padded.
// 64x64 block tile, 4 waves, each wave = 2x2 WMMA 16x16 tiles. Branchless
// b128 global->LDS staging; fragments read as contiguous 16B chunks.
// ---------------------------------------------------------------------------
#define GTM 64
#define GTN 64
#define GTK 32
#define LSTR (GTK + 8)           // LDS row stride in halfs (80B, 16B aligned)

__global__ __launch_bounds__(128)
void gemm_wmma_k(const _Float16* __restrict__ A, const _Float16* __restrict__ Bt,
                 const float* __restrict__ bias, float* __restrict__ C,
                 int M, int Npad, int Nreal, int Kpad)
{
    __shared__ _Float16 As[GTM][LSTR];
    __shared__ _Float16 Bs[GTN][LSTR];

    const int tid  = threadIdx.x;
    const int wave = tid >> 5;
    const int lane = tid & 31;
    const int half = lane >> 4;        // 0/1
    const int lm   = lane & 15;
    const int wm2  = wave >> 1;        // wave's 32-row group
    const int wn2  = wave & 1;         // wave's 32-col group

    const int m0 = blockIdx.y * GTM;
    const int n0 = blockIdx.x * GTN;

    const v8f vz = {0.f,0.f,0.f,0.f,0.f,0.f,0.f,0.f};
    v8f acc[2][2];
    acc[0][0] = vz; acc[0][1] = vz; acc[1][0] = vz; acc[1][1] = vz;

    // cooperative load map: thread t -> tile row t>>1, 16-half chunk (t&1)
    const int tr  = tid >> 1;
    const int tcb = (tid & 1) * 16;

    for (int k0 = 0; k0 < Kpad; k0 += GTK) {
        const _Float16* ag = A  + (size_t)(m0 + tr) * Kpad + k0 + tcb;
        const _Float16* bg = Bt + (size_t)(n0 + tr) * Kpad + k0 + tcb;
        *(v8h*)&As[tr][tcb]     = *(const v8h*)(ag);
        *(v8h*)&As[tr][tcb + 8] = *(const v8h*)(ag + 8);
        *(v8h*)&Bs[tr][tcb]     = *(const v8h*)(bg);
        *(v8h*)&Bs[tr][tcb + 8] = *(const v8h*)(bg + 8);
        __syncthreads();

        v16h afrag[2], bfrag[2];
        #pragma unroll
        for (int tm = 0; tm < 2; ++tm) {
            const _Float16* ar = &As[32 * wm2 + 16 * tm + lm][0];
            v8h a0 = *(const v8h*)(ar + 8 * half);        // K = 8h .. 8h+7
            v8h a1 = *(const v8h*)(ar + 16 + 8 * half);   // K = 16+8h ..
            afrag[tm] = __builtin_shufflevector(a0, a1,
                        0,1,2,3,4,5,6,7,8,9,10,11,12,13,14,15);
        }
        #pragma unroll
        for (int tn = 0; tn < 2; ++tn) {
            const _Float16* br = &Bs[32 * wn2 + 16 * tn + lm][0];
            v8h b0 = *(const v8h*)(br + 8 * half);        // K = 8h .. 8h+7
            v8h b1 = *(const v8h*)(br + 16 + 8 * half);
            bfrag[tn] = __builtin_shufflevector(b0, b1,
                        0,1,2,3,4,5,6,7,8,9,10,11,12,13,14,15);
        }
        #pragma unroll
        for (int tm = 0; tm < 2; ++tm)
            #pragma unroll
            for (int tn = 0; tn < 2; ++tn)
                acc[tm][tn] = __builtin_amdgcn_wmma_f32_16x16x32_f16(
                    false, afrag[tm], false, bfrag[tn],
                    (short)0, acc[tm][tn], false, false);
        __syncthreads();
    }

    #pragma unroll
    for (int tm = 0; tm < 2; ++tm)
        #pragma unroll
        for (int tn = 0; tn < 2; ++tn) {
            int col = n0 + 32 * wn2 + 16 * tn + lm;
            float bv = (col < Nreal) ? bias[col] : 0.f;
            #pragma unroll
            for (int r = 0; r < 8; ++r) {
                int row = m0 + 32 * wm2 + 16 * tm + 8 * half + r;
                C[(size_t)row * Npad + col] = acc[tm][tn][r] + bv;
            }
        }
}

// ---------------------------------------------------------------------------
// Weight pack: f32 B[K,N] -> f16 Bt[Npad,Kpad] (transposed, zero-filled)
// ---------------------------------------------------------------------------
__global__ __launch_bounds__(256)
void pack_bt_k(const float* __restrict__ B, _Float16* __restrict__ Bt,
               int K, int N, int Kpad, int Npad)
{
    size_t idx = (size_t)blockIdx.x * blockDim.x + threadIdx.x;
    if (idx >= (size_t)Npad * Kpad) return;
    int k = (int)(idx % Kpad);
    int n = (int)(idx / Kpad);
    float v = (n < N && k < K) ? B[(size_t)k * N + n] : 0.f;
    Bt[idx] = (_Float16)v;
}

// Activation pack: f32 [M,K] -> f16 [M,Kpad] zero-filled
__global__ __launch_bounds__(256)
void pack_a16_k(const float* __restrict__ in, _Float16* __restrict__ out,
                int M, int K, int Kpad)
{
    size_t idx = (size_t)blockIdx.x * blockDim.x + threadIdx.x;
    if (idx >= (size_t)M * Kpad) return;
    int k = (int)(idx % Kpad);
    int m = (int)(idx / Kpad);
    out[idx] = (_Float16)((k < K) ? in[(size_t)m * K + k] : 0.f);
}

// ---------------------------------------------------------------------------
// LayerNorm over `width` elems of a row with pitch `pitch`, + activation,
// in place (f32). Optionally also emits a zero-padded f16 copy (GEMM A form).
// ACT: 0=identity 1=relu 2=elu 3=tanh 4=sigmoid
// ---------------------------------------------------------------------------
template<int ACT>
__global__ __launch_bounds__(256)
void ln_act_k(float* __restrict__ buf, int width, int pitch,
              _Float16* __restrict__ out16, int pitch16)
{
    __shared__ float ssum[256];
    __shared__ float ssq[256];
    float* p = buf + (size_t)blockIdx.x * pitch;

    float s = 0.f, q = 0.f;
    for (int i = threadIdx.x; i < width; i += 256) { float v = p[i]; s += v; q += v * v; }
    ssum[threadIdx.x] = s; ssq[threadIdx.x] = q;
    __syncthreads();
    for (int off = 128; off > 0; off >>= 1) {
        if (threadIdx.x < off) {
            ssum[threadIdx.x] += ssum[threadIdx.x + off];
            ssq[threadIdx.x]  += ssq[threadIdx.x + off];
        }
        __syncthreads();
    }
    const float mean = ssum[0] / (float)width;
    const float var  = ssq[0] / (float)width - mean * mean;
    const float rs   = rsqrtf(var + EPSLN);

    for (int i = threadIdx.x; i < width; i += 256) {
        float v = (p[i] - mean) * rs;
        if (ACT == 1) v = v > 0.f ? v : 0.f;
        if (ACT == 2) v = v > 0.f ? v : (expf(v) - 1.f);
        if (ACT == 3) v = tanhf(v);
        if (ACT == 4) v = 1.f / (1.f + expf(-v));
        p[i] = v;
    }
    if (out16) {
        _Float16* o = out16 + (size_t)blockIdx.x * pitch16;
        for (int i = threadIdx.x; i < pitch16; i += 256)
            o[i] = (_Float16)((i < width) ? p[i] : 0.f);
    }
}

// new_state(pitch HP) -> xr16[384,HP] = f16(sigmoid(ln)), state_out = sigmoid
__global__ __launch_bounds__(256)
void post_out_k(const float* __restrict__ ns, _Float16* __restrict__ xr16,
                float* __restrict__ state_out)
{
    __shared__ float ssum[256];
    __shared__ float ssq[256];
    const float* p = ns + (size_t)blockIdx.x * HP;

    float s = 0.f, q = 0.f;
    for (int i = threadIdx.x; i < HDIM; i += 256) { float v = p[i]; s += v; q += v * v; }
    ssum[threadIdx.x] = s; ssq[threadIdx.x] = q;
    __syncthreads();
    for (int off = 128; off > 0; off >>= 1) {
        if (threadIdx.x < off) {
            ssum[threadIdx.x] += ssum[threadIdx.x + off];
            ssq[threadIdx.x]  += ssq[threadIdx.x + off];
        }
        __syncthreads();
    }
    const float mean = ssum[0] / (float)HDIM;
    const float var  = ssq[0] / (float)HDIM - mean * mean;
    const float rs   = rsqrtf(var + EPSLN);

    for (int i = threadIdx.x; i < HP; i += 256) {
        if (i < HDIM) {
            float v = p[i];
            xr16[(size_t)blockIdx.x * HP + i] =
                (_Float16)(1.f / (1.f + expf(-(v - mean) * rs)));
            state_out[(size_t)blockIdx.x * HDIM + i] = 1.f / (1.f + expf(-v));
        } else {
            xr16[(size_t)blockIdx.x * HP + i] = (_Float16)0.f;
        }
    }
}

// ---------------------------------------------------------------------------
// Encoder conv (channels-last, stride 2, pad 1, k=4; kx pairs with the FIRST
// spatial dim due to the reference's transpose sandwich).
// ---------------------------------------------------------------------------
__global__ __launch_bounds__(256)
void conv_enc_k(const float* __restrict__ in, const float* __restrict__ w,
                const float* __restrict__ bias, float* __restrict__ out,
                int S, int Cin, int Cout)
{
    const int O = S >> 1;
    size_t idx = (size_t)blockIdx.x * blockDim.x + threadIdx.x;
    size_t total = (size_t)BKROWS * O * O * Cout;
    if (idx >= total) return;
    int oc = (int)(idx % Cout);
    size_t t = idx / Cout;
    int q = (int)(t % O); t /= O;
    int p = (int)(t % O);
    int n = (int)(t / O);

    float acc = bias[oc];
    for (int kx = 0; kx < 4; ++kx) {
        int i1 = 2 * p - 1 + kx;
        if (i1 < 0 || i1 >= S) continue;
        for (int ky = 0; ky < 4; ++ky) {
            int i2 = 2 * q - 1 + ky;
            if (i2 < 0 || i2 >= S) continue;
            const float* ip = in + ((size_t)(n * S + i1) * S + i2) * Cin;
            const float* wp = w + (size_t)oc * Cin * 16 + ky * 4 + kx;
            for (int c = 0; c < Cin; ++c)
                acc += ip[c] * wp[c * 16];
        }
    }
    out[idx] = acc;
}

// Decoder conv with up4 folded in: i = (2p-1+k)>>2 on the original grid
__global__ __launch_bounds__(256)
void conv_dec_k(const float* __restrict__ in, const float* __restrict__ w,
                const float* __restrict__ bias, float* __restrict__ out,
                int S, int Cin, int Cout, int sigm)
{
    const int U = 4 * S;
    const int O = U >> 1;
    size_t idx = (size_t)blockIdx.x * blockDim.x + threadIdx.x;
    size_t total = (size_t)BKROWS * O * O * Cout;
    if (idx >= total) return;
    int oc = (int)(idx % Cout);
    size_t t = idx / Cout;
    int q = (int)(t % O); t /= O;
    int p = (int)(t % O);
    int n = (int)(t / O);

    float acc = bias[oc];
    for (int kx = 0; kx < 4; ++kx) {
        int i1u = 2 * p - 1 + kx;
        if (i1u < 0 || i1u >= U) continue;
        int i1 = i1u >> 2;
        for (int ky = 0; ky < 4; ++ky) {
            int i2u = 2 * q - 1 + ky;
            if (i2u < 0 || i2u >= U) continue;
            int i2 = i2u >> 2;
            const float* ip = in + ((size_t)(n * S + i1) * S + i2) * Cin;
            const float* wp = w + (size_t)oc * Cin * 16 + ky * 4 + kx;
            for (int c = 0; c < Cin; ++c)
                acc += ip[c] * wp[c * 16];
        }
    }
    out[idx] = sigm ? (1.f / (1.f + expf(-acc))) : acc;
}

// pair16[2688, PKP] f16: [0:250]=s1[b,i], [250:500]=s1[b, j<i?j:j+1], tail 0
__global__ __launch_bounds__(256)
void build_pair_k(const float* __restrict__ s1, _Float16* __restrict__ pair16)
{
    size_t idx = (size_t)blockIdx.x * blockDim.x + threadIdx.x;
    if (idx >= (size_t)PROWS * PKP) return;
    int c = (int)(idx % PKP);
    int r = (int)(idx / PKP);
    int b = r / 56, rem = r % 56;
    int i = rem / 7, j = rem % 7;
    float v = 0.f;
    if (c < HDIM) {
        v = s1[((size_t)b * KGRP + i) * HP + c];
    } else if (c < 2 * HDIM) {
        int src = (j < i) ? j : j + 1;
        v = s1[((size_t)b * KGRP + src) * HP + (c - HDIM)];
    }
    pair16[idx] = (_Float16)v;
}

// att = sigmoid(t @ att2_w + att2_b), t pitch A1P, N=1, K=100
__global__ __launch_bounds__(256)
void att2_k(const float* __restrict__ t, const float* __restrict__ w,
            const float* __restrict__ b, float* __restrict__ out)
{
    int r = blockIdx.x * blockDim.x + threadIdx.x;
    if (r >= PROWS) return;
    float acc = b[0];
    for (int j = 0; j < 100; ++j) acc += t[(size_t)r * A1P + j] * w[j];
    out[r] = 1.f / (1.f + expf(-acc));
}

// effect[bk,h] = sum_j ctx[bk*7+j, h] * att[bk*7+j]   (ctx pitch HP)
__global__ __launch_bounds__(256)
void effect_k(const float* __restrict__ ctx, const float* __restrict__ att,
              float* __restrict__ eff)
{
    size_t idx = (size_t)blockIdx.x * blockDim.x + threadIdx.x;
    if (idx >= (size_t)BKROWS * HDIM) return;
    int h = (int)(idx % HDIM);
    int row = (int)(idx / HDIM);
    float s = 0.f;
    for (int j = 0; j < 7; ++j)
        s += ctx[((size_t)row * 7 + j) * HP + h] * att[row * 7 + j];
    eff[idx] = s;
}

// tot16[384, TKP] f16 = concat([s1(250, pitch HP), eff(250), henc(512)]), tail 0
__global__ __launch_bounds__(256)
void concat_k(const float* __restrict__ s1, const float* __restrict__ eff,
              const float* __restrict__ henc, _Float16* __restrict__ tot16)
{
    size_t idx = (size_t)blockIdx.x * blockDim.x + threadIdx.x;
    if (idx >= (size_t)BKROWS * TKP) return;
    int c = (int)(idx % TKP);
    int r = (int)(idx / TKP);
    float v = 0.f;
    if (c < HDIM)          v = s1[(size_t)r * HP + c];
    else if (c < 2 * HDIM) v = eff[(size_t)r * HDIM + (c - HDIM)];
    else if (c < 1012)     v = henc[(size_t)r * 512 + (c - 2 * HDIM)];
    tot16[idx] = (_Float16)v;
}

// ---------------------------------------------------------------------------
extern "C" void kernel_launch(void* const* d_in, const int* in_sizes, int n_in,
                              void* d_out, int out_size, void* d_ws, size_t ws_size,
                              hipStream_t stream)
{
    (void)in_sizes; (void)n_in; (void)out_size; (void)ws_size;
    const float* x      = (const float*)d_in[0];
    const float* state  = (const float*)d_in[1];
    const float* c1w    = (const float*)d_in[2];  const float* c1b  = (const float*)d_in[3];
    const float* c2w    = (const float*)d_in[4];  const float* c2b  = (const float*)d_in[5];
    const float* c3w    = (const float*)d_in[6];  const float* c3b  = (const float*)d_in[7];
    const float* efc_w  = (const float*)d_in[8];  const float* efc_b  = (const float*)d_in[9];
    const float* renc_w = (const float*)d_in[10]; const float* renc_b = (const float*)d_in[11];
    const float* core_w = (const float*)d_in[12]; const float* core_b = (const float*)d_in[13];
    const float* ctx_w  = (const float*)d_in[14]; const float* ctx_b  = (const float*)d_in[15];
    const float* att1_w = (const float*)d_in[16]; const float* att1_b = (const float*)d_in[17];
    const float* att2_w = (const float*)d_in[18]; const float* att2_b = (const float*)d_in[19];
    const float* out_w  = (const float*)d_in[20]; const float* out_b  = (const float*)d_in[21];
    const float* dfc1_w = (const float*)d_in[22]; const float* dfc1_b = (const float*)d_in[23];
    const float* dfc2_w = (const float*)d_in[24]; const float* dfc2_b = (const float*)d_in[25];
    const float* d1w    = (const float*)d_in[26]; const float* d1b    = (const float*)d_in[27];
    const float* d2w    = (const float*)d_in[28]; const float* d2b    = (const float*)d_in[29];
    const float* d3w    = (const float*)d_in[30]; const float* d3b    = (const float*)d_in[31];

    // ---- workspace carve (256B aligned) ----
    char* base = (char*)d_ws;
    size_t off = 0;
    auto carve = [&](size_t bytes) -> char* {
        char* p = base + off;
        off += (bytes + 255) & ~(size_t)255;
        return p;
    };
    // f32 buffers
    float* h1   = (float*)carve((size_t)BKROWS * 32 * 32 * 16 * 4);   // also e2
    float* h2   = (float*)carve((size_t)BKROWS * 16 * 16 * 32 * 4);   // also e1
    float* h3   = (float*)carve((size_t)BKROWS * 4096 * 4);           // also dd2
    float* henc = (float*)carve((size_t)BKROWS * 512 * 4);
    float* s1   = (float*)carve((size_t)BKROWS * HP * 4);
    float* corb = (float*)carve((size_t)PROWS * HP * 4);
    float* ctxb = (float*)carve((size_t)PROWS * HP * 4);
    float* at1  = (float*)carve((size_t)PROWS * A1P * 4);
    float* at2  = (float*)carve((size_t)PROWS * 4);
    float* eff  = (float*)carve((size_t)BKROWS * HDIM * 4);
    float* nst  = (float*)carve((size_t)BKROWS * HP * 4);
    float* dd1  = (float*)carve((size_t)BKROWS * 512 * 4);
    // f16 activation buffers (GEMM A operands)
    _Float16* h3_16   = (_Float16*)carve((size_t)BKROWS * 4096 * 2);
    _Float16* state16 = (_Float16*)carve((size_t)BKROWS * HP * 2);
    _Float16* pair16  = (_Float16*)carve((size_t)PROWS * PKP * 2);
    _Float16* corb16  = (_Float16*)carve((size_t)PROWS * HP * 2);
    _Float16* tot16   = (_Float16*)carve((size_t)BKROWS * TKP * 2);
    _Float16* xr16    = (_Float16*)carve((size_t)BKROWS * HP * 2);
    _Float16* dd116   = (_Float16*)carve((size_t)BKROWS * 512 * 2);
    // f16 packed weights (Bt[Npad][Kpad])
    _Float16* efc_bt  = (_Float16*)carve((size_t)512 * 4096 * 2);
    _Float16* renc_bt = (_Float16*)carve((size_t)HP * HP * 2);
    _Float16* core_bt = (_Float16*)carve((size_t)HP * PKP * 2);
    _Float16* ctx_bt  = (_Float16*)carve((size_t)HP * HP * 2);
    _Float16* att1_bt = (_Float16*)carve((size_t)A1P * HP * 2);
    _Float16* out_bt  = (_Float16*)carve((size_t)HP * TKP * 2);
    _Float16* dfc1_bt = (_Float16*)carve((size_t)512 * HP * 2);
    _Float16* dfc2_bt = (_Float16*)carve((size_t)4096 * 512 * 2);
    float* dd2 = h3;
    float* e1  = h2;
    float* e2  = h1;

    float* x_out     = (float*)d_out;
    float* state_out = x_out + (size_t)BKROWS * 4096;

    auto packbt = [&](const float* B, _Float16* Bt, int K, int N, int Kp, int Np) {
        size_t tot = (size_t)Np * Kp;
        pack_bt_k<<<(unsigned)((tot + 255) / 256), 256, 0, stream>>>(B, Bt, K, N, Kp, Np);
    };
    auto gemm = [&](const _Float16* A, const _Float16* Bt, const float* bias,
                    float* C, int M, int Np, int Nreal, int Kp) {
        dim3 g(Np / GTN, M / GTM);
        gemm_wmma_k<<<g, dim3(128), 0, stream>>>(A, Bt, bias, C, M, Np, Nreal, Kp);
    };

    // ---- one-time weight packs (independent of data path) ----
    packbt(efc_w,  efc_bt,  4096, 512,  4096, 512);
    packbt(renc_w, renc_bt, HDIM, HDIM, HP,   HP);
    packbt(core_w, core_bt, 2*HDIM, HDIM, PKP, HP);
    packbt(ctx_w,  ctx_bt,  HDIM, HDIM, HP,   HP);
    packbt(att1_w, att1_bt, HDIM, 100,  HP,   A1P);
    packbt(out_w,  out_bt,  1012, HDIM, TKP,  HP);
    packbt(dfc1_w, dfc1_bt, HDIM, 512,  HP,   512);
    packbt(dfc2_w, dfc2_bt, 512,  4096, 512,  4096);
    {
        size_t tot = (size_t)BKROWS * HP;
        pack_a16_k<<<(unsigned)((tot + 255) / 256), 256, 0, stream>>>(state, state16,
                                                                      BKROWS, HDIM, HP);
    }

    // ---- encoder convs ----
    {
        size_t t1 = (size_t)BKROWS * 32 * 32 * 16;
        conv_enc_k<<<(unsigned)((t1 + 255) / 256), 256, 0, stream>>>(x, c1w, c1b, h1, 64, 1, 16);
        ln_act_k<2><<<BKROWS, 256, 0, stream>>>(h1, 32*32*16, 32*32*16, (_Float16*)nullptr, 0);
        size_t t2 = (size_t)BKROWS * 16 * 16 * 32;
        conv_enc_k<<<(unsigned)((t2 + 255) / 256), 256, 0, stream>>>(h1, c2w, c2b, h2, 32, 16, 32);
        ln_act_k<2><<<BKROWS, 256, 0, stream>>>(h2, 16*16*32, 16*16*32, (_Float16*)nullptr, 0);
        size_t t3 = (size_t)BKROWS * 4096;
        conv_enc_k<<<(unsigned)((t3 + 255) / 256), 256, 0, stream>>>(h2, c3w, c3b, h3, 16, 32, 64);
        ln_act_k<2><<<BKROWS, 256, 0, stream>>>(h3, 4096, 4096, h3_16, 4096);
    }

    // ---- dense encoder + relational core (WMMA GEMMs) ----
    gemm(h3_16, efc_bt, efc_b, henc, BKROWS, 512, 512, 4096);
    ln_act_k<2><<<BKROWS, 256, 0, stream>>>(henc, 512, 512, (_Float16*)nullptr, 0);

    gemm(state16, renc_bt, renc_b, s1, BKROWS, HP, HDIM, HP);
    ln_act_k<1><<<BKROWS, 256, 0, stream>>>(s1, HDIM, HP, (_Float16*)nullptr, 0);

    {
        size_t pt = (size_t)PROWS * PKP;
        build_pair_k<<<(unsigned)((pt + 255) / 256), 256, 0, stream>>>(s1, pair16);
    }
    gemm(pair16, core_bt, core_b, corb, PROWS, HP, HDIM, PKP);
    ln_act_k<1><<<PROWS, 256, 0, stream>>>(corb, HDIM, HP, corb16, HP);

    gemm(corb16, ctx_bt, ctx_b, ctxb, PROWS, HP, HDIM, HP);
    ln_act_k<1><<<PROWS, 256, 0, stream>>>(ctxb, HDIM, HP, (_Float16*)nullptr, 0);

    gemm(corb16, att1_bt, att1_b, at1, PROWS, A1P, 100, HP);
    ln_act_k<3><<<PROWS, 256, 0, stream>>>(at1, 100, A1P, (_Float16*)nullptr, 0);
    att2_k<<<(PROWS + 255) / 256, 256, 0, stream>>>(at1, att2_w, att2_b, at2);

    effect_k<<<(unsigned)(((size_t)BKROWS * HDIM + 255) / 256), 256, 0, stream>>>(ctxb, at2, eff);
    concat_k<<<(unsigned)(((size_t)BKROWS * TKP + 255) / 256), 256, 0, stream>>>(s1, eff, henc, tot16);

    gemm(tot16, out_bt, out_b, nst, BKROWS, HP, HDIM, TKP);
    post_out_k<<<BKROWS, 256, 0, stream>>>(nst, xr16, state_out);

    // ---- decoder ----
    gemm(xr16, dfc1_bt, dfc1_b, dd1, BKROWS, 512, 512, HP);
    ln_act_k<1><<<BKROWS, 256, 0, stream>>>(dd1, 512, 512, dd116, 512);

    gemm(dd116, dfc2_bt, dfc2_b, dd2, BKROWS, 4096, 4096, 512);
    ln_act_k<1><<<BKROWS, 256, 0, stream>>>(dd2, 4096, 4096, (_Float16*)nullptr, 0);

    {
        size_t u1 = (size_t)BKROWS * 16 * 16 * 32;
        conv_dec_k<<<(unsigned)((u1 + 255) / 256), 256, 0, stream>>>(dd2, d1w, d1b, e1, 8, 64, 32, 0);
        ln_act_k<1><<<BKROWS, 256, 0, stream>>>(e1, 16*16*32, 16*16*32, (_Float16*)nullptr, 0);
        size_t u2 = (size_t)BKROWS * 32 * 32 * 16;
        conv_dec_k<<<(unsigned)((u2 + 255) / 256), 256, 0, stream>>>(e1, d2w, d2b, e2, 16, 32, 16, 0);
        ln_act_k<1><<<BKROWS, 256, 0, stream>>>(e2, 32*32*16, 32*32*16, (_Float16*)nullptr, 0);
        size_t u3 = (size_t)BKROWS * 64 * 64;
        conv_dec_k<<<(unsigned)((u3 + 255) / 256), 256, 0, stream>>>(e2, d3w, d3b, x_out, 32, 16, 1, 1);
    }
}
